// CausalSelfAttention_1159641170519
// MI455X (gfx1250) — 16-bit and 8-bit matrix operand order repaired and checked by reading, compiled, not run on hardware
//
#include <hip/hip_runtime.h>
#include <hip/hip_bf16.h>


typedef __attribute__((ext_vector_type(8)))  int      v8i_t;
typedef __attribute__((ext_vector_type(8)))  float    v8f_t;
typedef __attribute__((ext_vector_type(16))) _Float16 v16h_t;

union HU { v16h_t h; int i[8]; };

#define B_SZ  2
#define T_LEN 2048
#define D_DIM 1024
#define BT    4096
#define H_CNT 16
#define HD    64
#define QEPS  1e-8f

__global__ void init_slots_kernel(float* slots) {
    if (threadIdx.x < 8) slots[threadIdx.x] = 0.0f;
}

__global__ void maxabs_kernel(const float* __restrict__ x, long long n,
                              float* __restrict__ slot) {
    float mx = 0.0f;
    long long stride = (long long)gridDim.x * blockDim.x;
    for (long long i = (long long)blockIdx.x * blockDim.x + threadIdx.x; i < n; i += stride)
        mx = fmaxf(mx, fabsf(x[i]));
    #pragma unroll
    for (int s = 16; s > 0; s >>= 1)
        mx = fmaxf(mx, __shfl_xor(mx, s, 32));
    if ((threadIdx.x & 31) == 0)
        atomicMax((unsigned int*)slot, __float_as_uint(mx));
}

__global__ void quantize_kernel(const float* __restrict__ x,
                                const float* __restrict__ slot,
                                signed char* __restrict__ q, long long n) {
    const float scale = fmaxf(slot[0] * (1.0f / 127.0f), QEPS);
    const float inv = 1.0f / scale;
    long long stride = (long long)gridDim.x * blockDim.x;
    for (long long i = (long long)blockIdx.x * blockDim.x + threadIdx.x; i < n; i += stride) {
        float r = rintf(x[i] * inv);
        r = fminf(fmaxf(r, -128.0f), 127.0f);
        q[i] = (signed char)(int)r;
    }
}

template <int MODE>
__global__ __launch_bounds__(128)
void gemm_i8_kernel(const signed char* __restrict__ Aq,
                    const signed char* __restrict__ Wq,
                    const float* __restrict__ bias,
                    const float* __restrict__ maA,
                    const float* __restrict__ maW,
                    void* __restrict__ outp) {
    __shared__ signed char Bs[64 * 64];

    const int tid  = threadIdx.x;
    const int lane = tid & 31;
    const int wid  = tid >> 5;
    const int mt   = blockIdx.x * 4 + wid;
    const int nb   = blockIdx.y * 64;
    const int half = lane >> 4;
    const int l15  = lane & 15;
    const int K    = D_DIM;

    v8i_t acc[4];
    #pragma unroll
    for (int nt = 0; nt < 4; ++nt)
        #pragma unroll
        for (int e = 0; e < 8; ++e) acc[nt][e] = 0;

    const long long arow = (long long)(mt * 16 + l15) * K;

    for (int k0 = 0; k0 < K; k0 += 64) {
        __syncthreads();
        #pragma unroll
        for (int i = 0; i < 2; ++i) {
            const int seg    = tid * 2 + i;
            const int nloc   = seg >> 2;
            const int kchunk = (seg & 3) * 16;
            const signed char* gp = Wq + (long long)(nb + nloc) * K + k0 + kchunk;
            const unsigned loff   = (unsigned)(uintptr_t)(Bs + nloc * 64 + kchunk);
            asm volatile("global_load_async_to_lds_b128 %0, %1, off"
                         :: "v"(loff), "v"(gp) : "memory");
        }
        asm volatile("s_wait_asynccnt 0x0" ::: "memory");
        __syncthreads();

        if (k0 + 64 < K)
            __builtin_prefetch(Aq + arow + k0 + 64, 0, 1);

        v8i_t a;
        #pragma unroll
        for (int vv = 0; vv < 8; ++vv) {
            const int koff = ((vv >> 1) & 1) * 16 + (vv >> 2) * 32 + half * 8 + (vv & 1) * 4;
            a[vv] = *(const int*)(Aq + arow + k0 + koff);
        }
        v8i_t b[4];
        #pragma unroll
        for (int nt = 0; nt < 4; ++nt) {
            const signed char* brow = Bs + (nt * 16 + l15) * 64;
            #pragma unroll
            for (int vv = 0; vv < 8; ++vv) {
                const int koff = ((vv >> 1) & 1) * 16 + (vv >> 2) * 32 + half * 8 + (vv & 1) * 4;
                b[nt][vv] = *(const int*)(brow + koff);
            }
        }
        #pragma unroll
        for (int nt = 0; nt < 4; ++nt)
            acc[nt] = __builtin_amdgcn_wmma_i32_16x16x64_iu8(
                true, a, true, b[nt], acc[nt], false, false);
    }

    const float sa = fmaxf(maA[0] * (1.0f / 127.0f), QEPS);
    const float sw = fmaxf(maW[0] * (1.0f / 127.0f), QEPS);
    const float cs = sa * sw;

    #pragma unroll
    for (int nt = 0; nt < 4; ++nt) {
        const int ncol = nb + nt * 16 + l15;
        const float bv = bias[ncol];
        #pragma unroll
        for (int r = 0; r < 8; ++r) {
            const int row = mt * 16 + half * 8 + r;
            const float f = (float)acc[nt][r] * cs + bv;
            const int bb = row >> 11;
            const int tt = row & (T_LEN - 1);
            const int hh = ncol >> 6;
            const int hd = ncol & 63;
            if (MODE == 0) {
                _Float16* o = (_Float16*)outp;
                o[(((long long)bb * H_CNT + hh) * T_LEN + tt) * HD + hd] = (_Float16)f;
            } else if (MODE == 2) {
                _Float16* o = (_Float16*)outp;
                o[(((long long)bb * H_CNT + hh) * HD + hd) * T_LEN + tt] = (_Float16)f;
            } else {
                float* o = (float*)outp;
                o[(long long)row * D_DIM + ncol] = f;
            }
        }
    }
}

__global__ __launch_bounds__(32)
void attn_kernel(const _Float16* __restrict__ Q,
                 const _Float16* __restrict__ Kk,
                 const _Float16* __restrict__ Vt,
                 float* __restrict__ Y) {
    const int lane = threadIdx.x;
    const int qt   = blockIdx.x;
    const int h    = blockIdx.y;
    const int b    = blockIdx.z;
    const int l15  = lane & 15;
    const int half = lane >> 4;

    const long long base = ((long long)(b * H_CNT + h)) * T_LEN * HD;
    const _Float16* qp  = Q  + base;
    const _Float16* kp  = Kk + base;
    const _Float16* vtp = Vt + base;

    HU qb[2];
    {
        const long long qrow = (long long)(qt * 16 + l15) * HD;
        #pragma unroll
        for (int c = 0; c < 2; ++c)
            #pragma unroll
            for (int vv = 0; vv < 8; ++vv)
                qb[c].i[vv] = *(const int*)(qp + qrow + c * 32 + half * 16 + 2 * vv);
    }

    v8f_t o[4];
    #pragma unroll
    for (int nt = 0; nt < 4; ++nt)
        #pragma unroll
        for (int e = 0; e < 8; ++e) o[nt][e] = 0.0f;
    float mrow = -__builtin_inff();
    float lrow = 0.0f;

    const int nkb = (qt * 16 + 15) / 32 + 1;
    for (int kb = 0; kb < nkb; ++kb) {
        HU ka[2][2];
        #pragma unroll
        for (int sub = 0; sub < 2; ++sub) {
            const long long krow = (long long)(kb * 32 + sub * 16 + l15) * HD;
            #pragma unroll
            for (int c = 0; c < 2; ++c)
                #pragma unroll
                for (int vv = 0; vv < 8; ++vv) {
                    const int ko = half * 16 + vv * 2;
                    ka[sub][c].i[vv] = *(const int*)(kp + krow + c * 32 + ko);
                }
        }
        v8f_t st[2];
        #pragma unroll
        for (int s = 0; s < 2; ++s)
            #pragma unroll
            for (int e = 0; e < 8; ++e) st[s][e] = 0.0f;
        #pragma unroll
        for (int sub = 0; sub < 2; ++sub)
            #pragma unroll
            for (int c = 0; c < 2; ++c)
                st[sub] = __builtin_amdgcn_wmma_f32_16x16x32_f16(
                    false, ka[sub][c].h, false, qb[c].h, (short)0, st[sub], false, false);

        float sv[16];
        float rmax = -__builtin_inff();
        const int qg = qt * 16 + l15;
        if (kb * 32 + 31 > qt * 16) {
            #pragma unroll
            for (int sub = 0; sub < 2; ++sub)
                #pragma unroll
                for (int r = 0; r < 8; ++r) {
                    const int keyg = kb * 32 + sub * 16 + half * 8 + r;
                    float xx = st[sub][r] * 0.125f;
                    if (keyg > qg) xx = -__builtin_inff();
                    sv[sub * 8 + r] = xx;
                    rmax = fmaxf(rmax, xx);
                }
        } else {
            #pragma unroll
            for (int sub = 0; sub < 2; ++sub)
                #pragma unroll
                for (int r = 0; r < 8; ++r) {
                    const float xx = st[sub][r] * 0.125f;
                    sv[sub * 8 + r] = xx;
                    rmax = fmaxf(rmax, xx);
                }
        }
        rmax = fmaxf(rmax, __shfl_xor(rmax, 16, 32));
        const float mnew  = fmaxf(mrow, rmax);
        const float alpha = __expf(mrow - mnew);

        HU pa;
        float psum = 0.0f;
        #pragma unroll
        for (int e = 0; e < 16; ++e) {
            const float p = __expf(sv[e] - mnew);
            psum += p;
            pa.h[e] = (_Float16)p;
        }
        psum += __shfl_xor(psum, 16, 32);
        lrow = lrow * alpha + psum;
        mrow = mnew;

        float alr[8];
        #pragma unroll
        for (int r = 0; r < 8; ++r) alr[r] = __shfl(alpha, half * 8 + r, 32);

        HU vb[4];
        #pragma unroll
        for (int nt = 0; nt < 4; ++nt) {
            const long long vrow = (long long)(nt * 16 + l15) * T_LEN
                                 + kb * 32 + half * 8;
            #pragma unroll
            for (int vv = 0; vv < 8; ++vv)
                vb[nt].i[vv] = *(const int*)(vtp + vrow + (vv >> 2) * 16 + (vv & 3) * 2);
        }
        #pragma unroll
        for (int nt = 0; nt < 4; ++nt) {
            #pragma unroll
            for (int r = 0; r < 8; ++r) o[nt][r] *= alr[r];
            o[nt] = __builtin_amdgcn_wmma_f32_16x16x32_f16(
                false, pa.h, false, vb[nt].h, (short)0, o[nt], false, false);
        }
    }

    float lr[8];
    #pragma unroll
    for (int r = 0; r < 8; ++r) lr[r] = __shfl(lrow, half * 8 + r, 32);
    #pragma unroll
    for (int nt = 0; nt < 4; ++nt)
        #pragma unroll
        for (int r = 0; r < 8; ++r) {
            const int t = qt * 16 + half * 8 + r;
            const long long yi = ((long long)b * T_LEN + t) * D_DIM
                               + h * HD + nt * 16 + l15;
            Y[yi] = o[nt][r] / lr[r];
        }
}

extern "C" void kernel_launch(void* const* d_in, const int* in_sizes, int n_in,
                              void* d_out, int out_size, void* d_ws, size_t ws_size,
                              hipStream_t stream) {
    (void)in_sizes; (void)n_in; (void)out_size; (void)ws_size;
    const float* x  = (const float*)d_in[0];
    const float* Wq = (const float*)d_in[1];
    const float* bq = (const float*)d_in[2];
    const float* Wk = (const float*)d_in[3];
    const float* bk = (const float*)d_in[4];
    const float* Wv = (const float*)d_in[5];
    const float* bv = (const float*)d_in[6];
    const float* Wo = (const float*)d_in[7];
    const float* bo = (const float*)d_in[8];

    const long long nX = (long long)BT * D_DIM;
    const long long nW = (long long)D_DIM * D_DIM;

    char* ws = (char*)d_ws;
    float*       ma  = (float*)ws;
    signed char* qx  = (signed char*)(ws + 256);
    signed char* qWq = qx  + nX;
    signed char* qWk = qWq + nW;
    signed char* qWv = qWk + nW;
    signed char* qWo = qWv + nW;
    _Float16*    qh  = (_Float16*)(qWo + nW);
    _Float16*    kh  = qh + nX;
    _Float16*    vth = kh + nX;
    float*       y   = (float*)(vth + nX);
    signed char* qy  = (signed char*)(y + nX);

    init_slots_kernel<<<1, 32, 0, stream>>>(ma);

    maxabs_kernel<<<512, 256, 0, stream>>>(x,  nX, ma + 0);
    maxabs_kernel<<<256, 256, 0, stream>>>(Wq, nW, ma + 1);
    maxabs_kernel<<<256, 256, 0, stream>>>(Wk, nW, ma + 2);
    maxabs_kernel<<<256, 256, 0, stream>>>(Wv, nW, ma + 3);
    maxabs_kernel<<<256, 256, 0, stream>>>(Wo, nW, ma + 4);

    quantize_kernel<<<512, 256, 0, stream>>>(x,  ma + 0, qx,  nX);
    quantize_kernel<<<256, 256, 0, stream>>>(Wq, ma + 1, qWq, nW);
    quantize_kernel<<<256, 256, 0, stream>>>(Wk, ma + 2, qWk, nW);
    quantize_kernel<<<256, 256, 0, stream>>>(Wv, ma + 3, qWv, nW);
    quantize_kernel<<<256, 256, 0, stream>>>(Wo, ma + 4, qWo, nW);

    dim3 gg(BT / 64, D_DIM / 64);
    gemm_i8_kernel<0><<<gg, 128, 0, stream>>>(qx, qWq, bq, ma + 0, ma + 1, (void*)qh);
    gemm_i8_kernel<0><<<gg, 128, 0, stream>>>(qx, qWk, bk, ma + 0, ma + 2, (void*)kh);
    gemm_i8_kernel<2><<<gg, 128, 0, stream>>>(qx, qWv, bv, ma + 0, ma + 3, (void*)vth);

    attn_kernel<<<dim3(T_LEN / 16, H_CNT, B_SZ), 32, 0, stream>>>(qh, kh, vth, y);

    maxabs_kernel<<<512, 256, 0, stream>>>(y, nX, ma + 5);
    quantize_kernel<<<512, 256, 0, stream>>>(y, ma + 5, qy, nX);

    gemm_i8_kernel<1><<<gg, 128, 0, stream>>>(qy, qWo, bo, ma + 5, ma + 4, d_out);
}
